// SwinTransBlock_34437047779508
// MI455X (gfx1250) — matrix operand order repaired and checked by reading, not compiled, not run on hardware
//
#include <hip/hip_runtime.h>

// ---------------- types / constants ----------------
typedef _Float16 v8h  __attribute__((ext_vector_type(8)));
typedef _Float16 v16h __attribute__((ext_vector_type(16)));
typedef float    v8f  __attribute__((ext_vector_type(8)));

#define CDIM   512
#define DHEAD  64
#define SQRT2F 1.4142135623730951f

// window token map: window w_=((b*8)+hb)*8+wb, pos n=r*8+c -> global token
static __device__ __forceinline__ int tok_index(int w_, int n) {
  int b = w_ >> 6, hb = (w_ >> 3) & 7, wb = w_ & 7;
  int r = n >> 3, c = n & 7;
  return (b << 12) + (((hb << 3) + r) << 6) + (wb << 3) + c;
}

static __device__ __forceinline__ v16h cat16(v8h lo, v8h hi) {
  return __builtin_shufflevector(lo, hi, 0,1,2,3,4,5,6,7,8,9,10,11,12,13,14,15);
}

// A operand (16x32 MxK, f16) from row-major A: lane<16 row M=lane, K {0..7,16..23}
static __device__ __forceinline__ v16h load_a_frag(const _Float16* A, int lda, int k0, int lane) {
  int hf = lane >> 4, m = lane & 15;
  const _Float16* p = A + (long)m * lda + k0 + hf * 8;
  v8h lo = *(const v8h*)p;
  v8h hi = *(const v8h*)(p + 16);
  return cat16(lo, hi);
}

// wave-relative LDS byte offset (low 32 bits of generic shared pointer)
static __device__ __forceinline__ unsigned lds_off(const void* p) {
  return (unsigned)(unsigned long long)(uintptr_t)p;
}

// CDNA5 async global->LDS copy, 16B per active lane (GVS addressing), tracked by ASYNCcnt
static __device__ __forceinline__ void async_copy_b128(unsigned ldsoff, unsigned goff,
                                                       const void* base) {
  asm volatile("global_load_async_to_lds_b128 %0, %1, %2"
               :: "v"(ldsoff), "v"(goff), "s"((unsigned long long)(uintptr_t)base)
               : "memory");
}
static __device__ __forceinline__ void wait_async0() {
  asm volatile("s_wait_asynccnt 0x0" ::: "memory");
}

// ---------------- weight convert (fold 1/sqrt(in) gain, fp32 -> f16) ----------------
__global__ __launch_bounds__(256) void wcvt_kernel(const float* __restrict__ s,
                                                   _Float16* __restrict__ d,
                                                   int n, float gain) {
  int i = blockIdx.x * 256 + threadIdx.x;
  if (i < n) d[i] = (_Float16)(s[i] * gain);
}

// ---------------- prep: x -> f16 (shortcut half of CAT) + L2-normalized f16 ----------------
__global__ __launch_bounds__(256) void prep_kernel(const float* __restrict__ x,
                                                   _Float16* __restrict__ cat,
                                                   _Float16* __restrict__ nrm) {
  int lane = threadIdx.x & 31, wid = threadIdx.x >> 5;
  long t = (long)blockIdx.x * 8 + wid;               // one wave per token
  const float* xp = x + t * CDIM;
  float xv[16]; float ss = 0.f;
#pragma unroll
  for (int j = 0; j < 16; ++j) { float v = xp[lane + j * 32]; xv[j] = v; ss += v * v; }
#pragma unroll
  for (int o = 16; o > 0; o >>= 1) ss += __shfl_xor(ss, o, 32);
  float scale = 1.f / fmaxf(sqrtf(ss), 1e-12f);      // F.normalize semantics
  _Float16* cp = cat + t * 1024;                     // CAT row stride 1024 (concat layout)
  _Float16* np = nrm + t * CDIM;
#pragma unroll
  for (int j = 0; j < 16; ++j) {
    int i = lane + j * 32;
    cp[i] = (_Float16)xv[j];
    np[i] = (_Float16)(xv[j] * scale);
  }
}

// ---------------- WMMA GEMM with async-LDS B staging ----------------
// Block: 256 thr = 8 waves, tile 256(M)x64(N); wave tile 32x64 (2 Mt x 4 Nt accs).
// Per k-step: block stages 64x32 f16 B panel to LDS via global_load_async_to_lds_b128
// (double buffered); waves read B frags from LDS, A frags direct from global.
// EPI 0: f16 out, *out_scale after bias.  EPI 1: leaky_relu(0.2)*sqrt2, f16 out.  EPI 2: f32 out.
#define BSTRIDE 48   // halfs per LDS B row (16B aligned, bank-spread)
template <int EPI>
__global__ __launch_bounds__(256) void gemm_kernel(const _Float16* __restrict__ A, int lda,
                                                   const _Float16* __restrict__ W, int ldw,
                                                   const float* __restrict__ bias,
                                                   void* __restrict__ out, int ldc,
                                                   int K, float out_scale) {
  __shared__ _Float16 Bs[2][64 * BSTRIDE];
  int tid  = threadIdx.x;
  int lane = tid & 31, wid = tid >> 5;
  int hf = lane >> 4, nl = lane & 15;
  int m0  = blockIdx.x * 256 + wid * 32;
  int n0b = blockIdx.y * 64;

  // staging role: thread copies 16B: B row n_l = tid>>2, k chunk (tid&3)*8
  int sn = tid >> 2, sk = (tid & 3) * 8;
  unsigned goff  = (unsigned)(((long)(n0b + sn) * ldw + sk) * sizeof(_Float16));
  unsigned sdst0 = lds_off(&Bs[0][sn * BSTRIDE + sk]);
  unsigned sdst1 = lds_off(&Bs[1][sn * BSTRIDE + sk]);

  async_copy_b128(sdst0, goff, W);                 // prefetch k-step 0 -> buf0
  const _Float16* A0 = A + (long)m0 * lda;
  v8f acc[2][4] = {};
  int nsteps = K >> 5;
  for (int s = 0; s < nsteps; ++s) {
    int k0 = s << 5;
    wait_async0();
    __syncthreads();                               // buf[s&1] ready for all waves
    if (s + 1 < nsteps)                            // prefetch next panel into other buffer
      async_copy_b128((s & 1) ? sdst0 : sdst1, goff + (unsigned)((k0 + 32) * sizeof(_Float16)), W);
    const _Float16* bs = &Bs[s & 1][0];
    v16h bf[4];
#pragma unroll
    for (int nt = 0; nt < 4; ++nt) {
      const _Float16* p = bs + (nt * 16 + nl) * BSTRIDE + hf * 8;
      bf[nt] = cat16(*(const v8h*)p, *(const v8h*)(p + 16));
    }
    v16h a0 = load_a_frag(A0,            lda, k0, lane);
    v16h a1 = load_a_frag(A0 + 16 * lda, lda, k0, lane);
#pragma unroll
    for (int nt = 0; nt < 4; ++nt) {
      acc[0][nt] = __builtin_amdgcn_wmma_f32_16x16x32_f16(false, a0, false, bf[nt], (short)0, acc[0][nt], false, false);
      acc[1][nt] = __builtin_amdgcn_wmma_f32_16x16x32_f16(false, a1, false, bf[nt], (short)0, acc[1][nt], false, false);
    }
  }
#pragma unroll
  for (int mt = 0; mt < 2; ++mt)
#pragma unroll
    for (int nt = 0; nt < 4; ++nt) {
      int col = n0b + nt * 16 + nl;
      float bv = bias[col];
#pragma unroll
      for (int r = 0; r < 8; ++r) {
        int row = m0 + mt * 16 + r + hf * 8;
        float v = acc[mt][nt][r] + bv;
        if (EPI == 1) v = (v >= 0.f ? v : 0.2f * v) * SQRT2F;
        else          v = v * out_scale;
        if (EPI == 2) ((float*)out)[(long)row * ldc + col] = v;
        else          ((_Float16*)out)[(long)row * ldc + col] = (_Float16)v;
      }
    }
}

// ---------------- windowed attention: one wave per (window, head) ----------------
__global__ __launch_bounds__(32) void attn_kernel(const _Float16* __restrict__ q16,  // pre-scaled by 1/8
                                                  const _Float16* __restrict__ k16,
                                                  const _Float16* __restrict__ v16,
                                                  const float*    __restrict__ mask,
                                                  _Float16* __restrict__ o16) {
  __shared__ float    S[64 * 72];     // logits (f32)
  __shared__ _Float16 P[64 * 72];     // softmax probs (f16), 144B row stride (16B aligned)
  __shared__ _Float16 Vt[64 * 72];    // V transposed: Vt[d][token]
  __shared__ float    kb[64];         // key-side mask bias
  int lane = threadIdx.x;
  int h  = blockIdx.x & 7;
  int w_ = blockIdx.x >> 3;
  int hf = lane >> 4, nl = lane & 15;

  // stage key-mask bias + V^T (global gather -> LDS transpose)
  for (int n = lane; n < 64; n += 32) {
    int g = tok_index(w_, n);
    kb[n] = (mask[g] == 0.f) ? -100.f : 0.f;
    const _Float16* vp = v16 + (long)g * CDIM + h * DHEAD;
    v8h rr[8];
#pragma unroll
    for (int j = 0; j < 8; ++j) rr[j] = *(const v8h*)(vp + j * 8);
#pragma unroll
    for (int j = 0; j < 8; ++j)
#pragma unroll
      for (int e = 0; e < 8; ++e) Vt[(j * 8 + e) * 72 + n] = rr[j][e];
  }
  __syncthreads();

  // S = (Q*0.125) K^T : preload all K fragments (B-operand, contiguous along d)
  v16h bk[4][2];
#pragma unroll
  for (int nt = 0; nt < 4; ++nt) {
    int g = tok_index(w_, nt * 16 + nl);
    const _Float16* kp = k16 + (long)g * CDIM + h * DHEAD;
#pragma unroll
    for (int ks = 0; ks < 2; ++ks) {
      const _Float16* p = kp + ks * 32 + hf * 8;
      bk[nt][ks] = cat16(*(const v8h*)p, *(const v8h*)(p + 16));
    }
  }
#pragma unroll
  for (int mt = 0; mt < 4; ++mt) {
    int g = tok_index(w_, mt * 16 + nl);
    const _Float16* qp = q16 + (long)g * CDIM + h * DHEAD;
    v16h a0, a1;
    {
      const _Float16* p0 = qp + 0  + hf * 8;
      const _Float16* p1 = qp + 32 + hf * 8;
      a0 = cat16(*(const v8h*)p0, *(const v8h*)(p0 + 16));
      a1 = cat16(*(const v8h*)p1, *(const v8h*)(p1 + 16));
    }
#pragma unroll
    for (int nt = 0; nt < 4; ++nt) {
      v8f acc = {};
      acc = __builtin_amdgcn_wmma_f32_16x16x32_f16(false, a0, false, bk[nt][0], (short)0, acc, false, false);
      acc = __builtin_amdgcn_wmma_f32_16x16x32_f16(false, a1, false, bk[nt][1], (short)0, acc, false, false);
      int col = nt * 16 + nl;
#pragma unroll
      for (int r = 0; r < 8; ++r) S[(mt * 16 + r + hf * 8) * 72 + col] = acc[r];
    }
  }
  __syncthreads();

  // softmax over keys (row-wise, f32) with key-mask bias; window mask am == 0 (shift=0)
#pragma unroll
  for (int rp = 0; rp < 2; ++rp) {
    int row = lane + rp * 32;
    float mx = -1e30f;
    for (int j = 0; j < 64; ++j) { float s = S[row * 72 + j] + kb[j]; S[row * 72 + j] = s; if (s > mx) mx = s; }
    float sum = 0.f;
    for (int j = 0; j < 64; ++j) { float e = __expf(S[row * 72 + j] - mx); S[row * 72 + j] = e; sum += e; }
    float rs = 1.f / sum;
    for (int j = 0; j < 64; ++j) P[row * 72 + j] = (_Float16)(S[row * 72 + j] * rs);
  }
  __syncthreads();

  // O = P V  (A from LDS P, B from LDS Vt), scatter-store to token-major o16
#pragma unroll
  for (int mt = 0; mt < 4; ++mt) {
    v16h a[2];
#pragma unroll
    for (int ks = 0; ks < 2; ++ks) {
      const _Float16* p = &P[(mt * 16 + nl) * 72 + ks * 32 + hf * 8];
      a[ks] = cat16(*(const v8h*)p, *(const v8h*)(p + 16));
    }
#pragma unroll
    for (int nt = 0; nt < 4; ++nt) {
      v8f acc = {};
#pragma unroll
      for (int ks = 0; ks < 2; ++ks) {
        const _Float16* p = &Vt[(nt * 16 + nl) * 72 + ks * 32 + hf * 8];
        v16h b = cat16(*(const v8h*)p, *(const v8h*)(p + 16));
        acc = __builtin_amdgcn_wmma_f32_16x16x32_f16(false, a[ks], false, b, (short)0, acc, false, false);
      }
      int dd = h * DHEAD + nt * 16 + nl;
#pragma unroll
      for (int r = 0; r < 8; ++r) {
        int g = tok_index(w_, mt * 16 + r + hf * 8);
        o16[(long)g * CDIM + dd] = (_Float16)acc[r];
      }
    }
  }
}

// ---------------- mo: per-window clip(sum(mask),0,1) broadcast ----------------
__global__ __launch_bounds__(32) void mo_kernel(const float* __restrict__ mask, float* __restrict__ mo) {
  int lane = threadIdx.x, w_ = blockIdx.x;
  int g0 = tok_index(w_, lane), g1 = tok_index(w_, lane + 32);
  float s = mask[g0] + mask[g1];
#pragma unroll
  for (int o = 16; o > 0; o >>= 1) s += __shfl_xor(s, o, 32);
  float m = fminf(fmaxf(s, 0.f), 1.f);
  mo[g0] = m; mo[g1] = m;
}

// ---------------- host launch ----------------
extern "C" void kernel_launch(void* const* d_in, const int* in_sizes, int n_in,
                              void* d_out, int out_size, void* d_ws, size_t ws_size,
                              hipStream_t stream) {
  const float* x    = (const float*)d_in[0];
  const float* mask = (const float*)d_in[1];
  const float* wq = (const float*)d_in[4];  const float* bq = (const float*)d_in[5];
  const float* wk = (const float*)d_in[6];  const float* bk = (const float*)d_in[7];
  const float* wv = (const float*)d_in[8];  const float* bv = (const float*)d_in[9];
  const float* wp = (const float*)d_in[10]; const float* bp = (const float*)d_in[11];
  const float* wf = (const float*)d_in[12]; const float* bf = (const float*)d_in[13];
  const float* w1 = (const float*)d_in[14]; const float* b1 = (const float*)d_in[15];
  const float* w2 = (const float*)d_in[16]; const float* b2 = (const float*)d_in[17];

  char* ws = (char*)d_ws;
  // f16 weights (gain folded)
  _Float16* WQ = (_Float16*)(ws + 0);
  _Float16* WK = (_Float16*)(ws + 524288);
  _Float16* WV = (_Float16*)(ws + 1048576);
  _Float16* WP = (_Float16*)(ws + 1572864);
  _Float16* WF = (_Float16*)(ws + 2097152);   // 512x1024
  _Float16* W1 = (_Float16*)(ws + 3145728);   // 1024x512
  _Float16* W2 = (_Float16*)(ws + 4194304);   // 512x1024
  // activations (token-major, f16)
  _Float16* CAT = (_Float16*)(ws + 8388608);    // [32768 x 1024]: cols 0-511 shortcut, 512-1023 xo
  _Float16* NRM = (_Float16*)(ws + 75497472);   // [32768 x 512] normalized x; reused as fuse out
  _Float16* Q16 = (_Float16*)(ws + 109051904);  // later reused (with K16) as fc1 out [32768x1024]
  _Float16* K16 = (_Float16*)(ws + 142606336);
  _Float16* V16 = (_Float16*)(ws + 176160768);
  _Float16* O16 = (_Float16*)(ws + 209715200);
  _Float16* YF  = NRM;                          // fuse output [32768 x 512]
  _Float16* H1  = Q16;                          // fc1 output [32768 x 1024] (spans Q16+K16)

  float* y_out  = (float*)d_out;                // [8*4096*512]
  float* mo_out = (float*)d_out + 16777216;     // [8*4096]

  const float g512  = 0.044194173824159216f;    // 1/sqrt(512)
  const float g1024 = 0.03125f;                 // 1/sqrt(1024)
  wcvt_kernel<<<dim3(1024), dim3(256), 0, stream>>>(wq, WQ, 262144, g512);
  wcvt_kernel<<<dim3(1024), dim3(256), 0, stream>>>(wk, WK, 262144, g512);
  wcvt_kernel<<<dim3(1024), dim3(256), 0, stream>>>(wv, WV, 262144, g512);
  wcvt_kernel<<<dim3(1024), dim3(256), 0, stream>>>(wp, WP, 262144, g512);
  wcvt_kernel<<<dim3(2048), dim3(256), 0, stream>>>(wf, WF, 524288, g1024);
  wcvt_kernel<<<dim3(2048), dim3(256), 0, stream>>>(w1, W1, 524288, g512);
  wcvt_kernel<<<dim3(2048), dim3(256), 0, stream>>>(w2, W2, 524288, g1024);

  prep_kernel<<<dim3(4096), dim3(256), 0, stream>>>(x, CAT, NRM);

  dim3 blk(256);
  dim3 g8(128, 8), g16(128, 16);   // M/256 blocks in x, N/64 in y
  // Q (scale by D^-0.5 = 1/8 in epilogue), K, V projections
  gemm_kernel<0><<<g8, blk, 0, stream>>>(NRM, 512, WQ, 512, bq, Q16, 512, 512, 0.125f);
  gemm_kernel<0><<<g8, blk, 0, stream>>>(NRM, 512, WK, 512, bk, K16, 512, 512, 1.0f);
  gemm_kernel<0><<<g8, blk, 0, stream>>>(CAT, 1024, WV, 512, bv, V16, 512, 512, 1.0f);

  // windowed attention: 512 windows x 8 heads, one wave each
  attn_kernel<<<dim3(4096), dim3(32), 0, stream>>>(Q16, K16, V16, mask, O16);

  // output projection -> second half of CAT (concat with shortcut)
  gemm_kernel<0><<<g8, blk, 0, stream>>>(O16, 512, WP, 512, bp, CAT + 512, 1024, 512, 1.0f);
  // fuse: [x, xo] (1024) -> 512, lrelu*sqrt2
  gemm_kernel<1><<<g8, blk, 0, stream>>>(CAT, 1024, WF, 1024, bf, YF, 512, 1024, 1.0f);
  // mlp fc1: 512 -> 1024, lrelu*sqrt2
  gemm_kernel<1><<<g16, blk, 0, stream>>>(YF, 512, W1, 512, b1, H1, 1024, 512, 1.0f);
  // mlp fc2: 1024 -> 512, f32 final output
  gemm_kernel<2><<<g8, blk, 0, stream>>>(H1, 1024, W2, 1024, b2, y_out, 512, 1024, 1.0f);

  // mask output
  mo_kernel<<<dim3(512), dim3(32), 0, stream>>>(mask, mo_out);
  (void)in_sizes; (void)n_in; (void)out_size; (void)ws_size;
}
